// Longformer_12240656794360
// MI455X (gfx1250) — compile-verified, matrix operand order checked by reading, not run on hardware
//
#include <hip/hip_runtime.h>
#include <hip/hip_bf16.h>
#include <math.h>

// ---------------- geometry ----------------
#define SQ   4096          // sequence length S
#define DM   768
#define NH   12
#define HD   64
#define NLAY 2
#define DFF  3072
#define NG   16            // G global tokens
#define WW   256           // window W
#define NCH  16            // S / W
#define TW   768           // 3*W
#define SP   4608          // S + 2*W
#define NCLS 3
#define NEGV (-1.0e9f)

typedef __attribute__((ext_vector_type(16))) __bf16 v16bf;
typedef __attribute__((ext_vector_type(8)))  float  v8f;
typedef __attribute__((ext_vector_type(4)))  unsigned tdm_g0_t;
typedef __attribute__((ext_vector_type(8)))  unsigned tdm_g1_t;

// ---------------- CDNA5 async copy helpers ----------------
// global_load_async_to_lds_b128: VDST = LDS byte offset (VGPR), VADDR = 64-bit
// global address (VGPR pair), no SADDR ("off"). Tracked by ASYNCcnt.
__device__ __forceinline__ void async_cp16(const __bf16* g, const __bf16* lds) {
  unsigned loff = (unsigned)(unsigned long long)(const void*)lds;  // addr[31:0] = LDS offset
  unsigned long long ga = (unsigned long long)g;
  asm volatile("global_load_async_to_lds_b128 %0, %1, off"
               :: "v"(loff), "v"(ga)
               : "memory");
}
__device__ __forceinline__ void wait_async0() {
  asm volatile("s_wait_asynccnt 0" ::: "memory");
}

// ---------------- CDNA5 Tensor Data Mover (TDM) ----------------
// 2D tile DMA global->LDS. D# per CDNA5 ISA:
//  group0: [1:0]=count=1, [63:32]=lds_addr, [120:64]=global_addr, [127:126]=type=2
//  group1: [15:0]=wg_mask=0, [17:16]=data_size (1 => 2B), [79:48]=tensor_dim0,
//          [111:80]=tensor_dim1, [127:112]=tile_dim0, [143:128]=tile_dim1,
//          [207:160]=tensor_dim0_stride. Tracked by TENSORcnt; EXEC ignored
//          (one DMA issued per executing wave).
__device__ __forceinline__ void tdm_load_2d(unsigned lds_off, const void* gaddr,
                                            unsigned rowElems, unsigned rows,
                                            unsigned strideElems) {
  unsigned long long ga = (unsigned long long)gaddr;
  tdm_g0_t g0;
  g0[0] = 1u;                                    // count=1 (user mode)
  g0[1] = lds_off;                               // LDS byte address
  g0[2] = (unsigned)ga;                          // global_addr[31:0]
  g0[3] = (unsigned)(ga >> 32) | (2u << 30);     // global_addr[56:32] | type=2
  tdm_g1_t g1;
  g1[0] = (1u << 16);                            // data_size=1 (2 bytes)
  g1[1] = (rowElems & 0xFFFFu) << 16;            // tensor_dim0[15:0]
  g1[2] = (rowElems >> 16) | ((rows & 0xFFFFu) << 16);  // dim0[31:16]|dim1[15:0]
  g1[3] = (rows >> 16) | ((rowElems & 0xFFFFu) << 16);  // dim1[31:16]|tile_dim0
  g1[4] = rows & 0xFFFFu;                        // tile_dim1 (tile_dim2=0)
  g1[5] = strideElems;                           // tensor_dim0_stride[31:0]
  g1[6] = 0u;
  g1[7] = 0u;
  asm volatile("tensor_load_to_lds %0, %1" :: "s"(g0), "s"(g1) : "memory");
}
__device__ __forceinline__ void wait_tensor0() {
  __builtin_amdgcn_s_wait_tensorcnt(0);
}

// ---------------- WMMA wave tile (generic, per-wave 16x16) ----------------
// C = A[M,K] * Bt[N,K]^T.  16-bit fragment layout per CDNA5 ISA: lane%16 = M/N,
// lane/16 = K-half; elems 0..7 <- K = kk+half*8+{0..7}, 8..15 <- +16.
template <typename AT>
__device__ inline v8f wmma_tile(const AT* __restrict__ A, int lda,
                                const __bf16* __restrict__ Bt, int ldb,
                                int K, int mrow, int ncol, int half) {
  v8f c = {0.f, 0.f, 0.f, 0.f, 0.f, 0.f, 0.f, 0.f};
  const AT*     ap0 = A  + (long)mrow * lda;
  const __bf16* bp0 = Bt + (long)ncol * ldb;
  for (int kk = 0; kk < K; kk += 32) {
    v16bf a, b;
    const AT*     ap = ap0 + kk + half * 8;
    const __bf16* bp = bp0 + kk + half * 8;
#pragma unroll
    for (int t = 0; t < 8; ++t) { a[t]     = (__bf16)ap[t];      b[t]     = bp[t]; }
#pragma unroll
    for (int t = 0; t < 8; ++t) { a[8 + t] = (__bf16)ap[16 + t]; b[8 + t] = bp[16 + t]; }
    c = __builtin_amdgcn_wmma_f32_16x16x32_bf16(false, a, false, b, (short)0, c,
                                                false, false);
  }
  return c;
}

// ------------- epilogue helper (shared semantics) -------------
__device__ __forceinline__ void store_elem(int mode, float v, long m, int n,
                                           int N, float* Cf, const float* resid,
                                           __bf16* Cb, long hs, int rs, int cs,
                                           int moff) {
  if (mode == 0) {
    long idx = m * N + n;
    Cf[idx] = v + (resid ? resid[idx] : 0.f);
  } else if (mode == 1) {
    int h = n / HD, d = n % HD;
    Cb[(long)h * hs + (m + moff) * (long)rs + (long)d * cs] = (__bf16)v;
  } else {
    Cb[m * N + n] = (__bf16)v;
  }
}

// ---------------- generic activation GEMM (small shapes, e.g. Qg) ----------
__global__ __launch_bounds__(256) void gemm_act(
    const __bf16* __restrict__ A, const __bf16* __restrict__ Bt,
    const float* __restrict__ bias, int M, int N, int K, int mode, float scale,
    int gelu, float* __restrict__ Cf, const float* __restrict__ resid,
    __bf16* __restrict__ Cb, long hs, int rs, int cs, int moff) {
  int lane = threadIdx.x & 31;
  int wave = threadIdx.x >> 5;
  int tileN = blockIdx.x;
  int tileM = blockIdx.y * 8 + wave;
  if (tileM * 16 >= M) return;  // wave-uniform
  int half = lane >> 4, l16 = lane & 15;
  v8f c = wmma_tile<__bf16>(A, K, Bt, K, K, tileM * 16 + l16, tileN * 16 + l16, half);
  int n = tileN * 16 + l16;
  float bv = bias ? bias[n] : 0.f;
#pragma unroll
  for (int r = 0; r < 8; ++r) {
    long m = tileM * 16 + half * 8 + r;
    float v = (c[r] + bv) * scale;
    if (gelu) v = 0.5f * v * (1.f + erff(v * 0.70710678118654752f));
    store_elem(mode, v, m, n, N, Cf, resid, Cb, hs, rs, cs, moff);
  }
}

// ---------------- blocked activation GEMM with TDM + async LDS staging ------
// Block tile 128x128, K-step 32. 8 waves as 2(M) x 4(N); each wave: 64x32
// register tile = 4x2 accumulators, 8 WMMAs per K-step from 6 LDS fragments.
// A tile: one TDM tensor_load_to_lds per wave (16-row slice), TENSORcnt.
// B tile: per-lane global_load_async_to_lds_b128, ASYNCcnt.
#define BLK_M 128
#define BLK_N 128
#define BLK_K 32
__global__ __launch_bounds__(256) void gemm_act_blk(
    const __bf16* __restrict__ A, const __bf16* __restrict__ Bt,
    const float* __restrict__ bias, int M, int N, int K, int mode, float scale,
    int gelu, float* __restrict__ Cf, const float* __restrict__ resid,
    __bf16* __restrict__ Cb, long hs, int rs, int cs, int moff) {
  __shared__ __align__(16) __bf16 sA[BLK_M * BLK_K];  // 8 KB
  __shared__ __align__(16) __bf16 sB[BLK_N * BLK_K];  // 8 KB
  int tid = threadIdx.x;
  int lane = tid & 31;
  int wv = __builtin_amdgcn_readfirstlane(tid >> 5);  // scalar wave id 0..7
  int wm = wv >> 2, wn = wv & 3;                      // 2 x 4 wave grid
  int half = lane >> 4, l16 = lane & 15;
  long bM = (long)blockIdx.y * BLK_M;
  long bN = (long)blockIdx.x * BLK_N;
  unsigned sA_base = (unsigned)(unsigned long long)(const void*)sA;
  const v8f vzero = {0.f, 0.f, 0.f, 0.f, 0.f, 0.f, 0.f, 0.f};
  v8f acc[4][2];
#pragma unroll
  for (int mi = 0; mi < 4; ++mi)
#pragma unroll
    for (int ni = 0; ni < 2; ++ni) acc[mi][ni] = vzero;

  for (int kk = 0; kk < K; kk += BLK_K) {
    __syncthreads();  // previous iteration's LDS readers done
    // A tile: each wave TDM-loads its 16x32 slice (row stride K elements)
    tdm_load_2d(sA_base + (unsigned)(wv * 16 * BLK_K * 2),
                A + (bM + wv * 16) * K + kk, BLK_K, 16, (unsigned)K);
    // B tile: 512 chunks of 16B, 2 per thread, async to LDS
#pragma unroll
    for (int u = 0; u < 2; ++u) {
      int c = tid + 256 * u;                // 0..511
      int row = c >> 2, col = (c & 3) * 8;  // 4 x 16B chunks per row
      async_cp16(Bt + (bN + row) * K + kk + col, &sB[row * BLK_K + col]);
    }
    wait_tensor0();
    wait_async0();
    __syncthreads();  // tiles visible to all waves
    v16bf af[4], bfr[2];
#pragma unroll
    for (int mi = 0; mi < 4; ++mi) {
      const __bf16* pr = &sA[(wm * 64 + mi * 16 + l16) * BLK_K + half * 8];
#pragma unroll
      for (int t = 0; t < 8; ++t) { af[mi][t] = pr[t]; af[mi][8 + t] = pr[16 + t]; }
    }
#pragma unroll
    for (int ni = 0; ni < 2; ++ni) {
      const __bf16* pr = &sB[(wn * 32 + ni * 16 + l16) * BLK_K + half * 8];
#pragma unroll
      for (int t = 0; t < 8; ++t) { bfr[ni][t] = pr[t]; bfr[ni][8 + t] = pr[16 + t]; }
    }
#pragma unroll
    for (int mi = 0; mi < 4; ++mi)
#pragma unroll
      for (int ni = 0; ni < 2; ++ni)
        acc[mi][ni] = __builtin_amdgcn_wmma_f32_16x16x32_bf16(
            false, af[mi], false, bfr[ni], (short)0, acc[mi][ni], false, false);
  }
  // epilogue
#pragma unroll
  for (int mi = 0; mi < 4; ++mi)
#pragma unroll
    for (int ni = 0; ni < 2; ++ni) {
      int n = (int)(bN + wn * 32 + ni * 16 + l16);
      float bv = bias ? bias[n] : 0.f;
#pragma unroll
      for (int r = 0; r < 8; ++r) {
        long m = bM + wm * 64 + mi * 16 + half * 8 + r;
        float v = (acc[mi][ni][r] + bv) * scale;
        if (gelu) v = 0.5f * v * (1.f + erff(v * 0.70710678118654752f));
        store_elem(mode, v, m, n, N, Cf, resid, Cb, hs, rs, cs, moff);
      }
    }
}

// ---------------- batched attention GEMM (z = head or head*chunk) -----------
template <typename AT>
__global__ __launch_bounds__(256) void gemm_z(
    const AT* __restrict__ A, const __bf16* __restrict__ Bt,
    float* __restrict__ C, int M, int N, int K, int lda, int ldb, int ldc,
    int zdiv, long hsA, long nsA, long hsB, long nsB, long hsC, long nsC,
    int acc) {
  int z = blockIdx.z;
  int zh = z / zdiv, zn = z % zdiv;
  const AT*     Az = A + zh * hsA + zn * nsA;
  const __bf16* Bz = Bt + zh * hsB + zn * nsB;
  float*        Cz = C + zh * hsC + zn * nsC;
  int lane = threadIdx.x & 31;
  int wave = threadIdx.x >> 5;
  int tileN = blockIdx.x;
  int tileM = blockIdx.y * 8 + wave;
  if (tileM * 16 >= M) return;
  int half = lane >> 4, l16 = lane & 15;
  v8f c = wmma_tile<AT>(Az, lda, Bz, ldb, K, tileM * 16 + l16, tileN * 16 + l16, half);
  int n = tileN * 16 + l16;
#pragma unroll
  for (int r = 0; r < 8; ++r) {
    int m = tileM * 16 + half * 8 + r;
    long idx = (long)m * ldc + n;
    if (acc) Cz[idx] += c[r]; else Cz[idx] = c[r];
  }
}

// ---------------- small utility kernels ----------------
__global__ void zero_bf16_k(__bf16* p, long n) {
  long i = (long)blockIdx.x * blockDim.x + threadIdx.x;
  long st = (long)gridDim.x * blockDim.x;
  for (; i < n; i += st) p[i] = (__bf16)0.f;
}

__global__ void transpose_w(const float* __restrict__ w, __bf16* __restrict__ wt,
                            int K, int N) {
  long tid = (long)blockIdx.x * blockDim.x + threadIdx.x;
  long total = (long)K * N;
  if (tid >= total) return;
  long n = tid / K, k = tid % K;
  wt[tid] = (__bf16)w[k * (long)N + n];
}

__global__ void embed_sum(const int* __restrict__ ids,
                          const float* __restrict__ ew,
                          const float* __restrict__ ep, float* __restrict__ y) {
  long tid = (long)blockIdx.x * blockDim.x + threadIdx.x;
  if (tid >= (long)SQ * DM) return;
  long s = tid / DM, c = tid % DM;
  y[tid] = ew[(long)ids[s] * DM + c] + ep[tid];
}

__device__ inline float blk_rmax(float v, float* red) {
  int t = threadIdx.x;
  red[t] = v; __syncthreads();
  for (int s = 128; s > 0; s >>= 1) {
    if (t < s) red[t] = fmaxf(red[t], red[t + s]);
    __syncthreads();
  }
  float r = red[0]; __syncthreads();
  return r;
}
__device__ inline float blk_rsum(float v, float* red) {
  int t = threadIdx.x;
  red[t] = v; __syncthreads();
  for (int s = 128; s > 0; s >>= 1) {
    if (t < s) red[t] += red[t + s];
    __syncthreads();
  }
  float r = red[0]; __syncthreads();
  return r;
}

// row LayerNorm over DM=768, blockDim=256, one row per block
__global__ __launch_bounds__(256) void ln_rows(const float* __restrict__ in,
                                               const float* __restrict__ gs,
                                               const float* __restrict__ gb,
                                               float* __restrict__ x,
                                               __bf16* __restrict__ xb) {
  __shared__ float red[256];
  int s = blockIdx.x, t = threadIdx.x;
  const float* row = in + (long)s * DM;
  float v[3], ls = 0.f;
#pragma unroll
  for (int u = 0; u < 3; ++u) { v[u] = row[t + u * 256]; ls += v[u]; }
  float mean = blk_rsum(ls, red) * (1.f / DM);
  ls = 0.f;
#pragma unroll
  for (int u = 0; u < 3; ++u) { float d = v[u] - mean; ls += d * d; }
  float var = blk_rsum(ls, red) * (1.f / DM);
  float rstd = rsqrtf(var + 1e-5f);
#pragma unroll
  for (int u = 0; u < 3; ++u) {
    int c = t + u * 256;
    float o = (v[u] - mean) * rstd * gs[c] + gb[c];
    x[(long)s * DM + c] = o;
    xb[(long)s * DM + c] = (__bf16)o;
  }
}

// band softmax row (h,q): mask band, add G global-key logits, softmax over
// G+3W, write band probs in place (fp32), write global-key V contribution
// into attn (band PV then accumulates on top).
__global__ __launch_bounds__(256) void band_softmax(
    const __bf16* __restrict__ qb, const __bf16* __restrict__ kpad,
    const __bf16* __restrict__ vt, const int* __restrict__ am,
    const int* __restrict__ gm, float* __restrict__ band,
    float* __restrict__ attn) {
  __shared__ float red[256];
  __shared__ float sPg[NG];
  int row = blockIdx.x;            // h*S + q
  int h = row / SQ, q = row % SQ;
  int n = q / WW, i = q % WW;
  int t = threadIdx.x;
  float* brow = band + (long)row * TW;
  float mv[3], lmax = -1e30f;
#pragma unroll
  for (int u = 0; u < 3; ++u) {
    int j = t + u * 256;
    int kpos = n * WW + j - WW;
    float v = brow[j];
    bool ok = (j >= i) && (j <= i + 2 * WW) && (kpos >= 0) && (kpos < SQ) &&
              (am[kpos] != 0) && (gm[kpos] == 0);
    v = ok ? v : NEGV;
    mv[u] = v;
    lmax = fmaxf(lmax, v);
  }
  float sgv = -1e30f;
  if (t < NG) {
    float acc = 0.f;
    const __bf16* qp = qb + ((long)h * SQ + q) * HD;
    const __bf16* kp = kpad + ((long)h * SP + WW + t) * HD;
#pragma unroll
    for (int d = 0; d < HD; ++d) acc += (float)qp[d] * (float)kp[d];
    sgv = (am[t] != 0) ? acc : NEGV;
    lmax = fmaxf(lmax, sgv);
  }
  float mx = blk_rmax(lmax, red);
  float ls = 0.f;
#pragma unroll
  for (int u = 0; u < 3; ++u) { mv[u] = __expf(mv[u] - mx); ls += mv[u]; }
  float esg = 0.f;
  if (t < NG) { esg = __expf(sgv - mx); ls += esg; }
  float sm = blk_rsum(ls, red);
  float inv = 1.f / sm;
#pragma unroll
  for (int u = 0; u < 3; ++u) brow[t + u * 256] = mv[u] * inv;
  if (t < NG) sPg[t] = esg * inv;
  __syncthreads();
  if (t < HD) {
    float acc = 0.f;
#pragma unroll
    for (int g = 0; g < NG; ++g)
      acc += sPg[g] * (float)vt[((long)h * HD + t) * SP + WW + g];
    attn[((long)h * SQ + q) * HD + t] = acc;
  }
}

// full softmax over S for the NG global-query rows, in place
__global__ __launch_bounds__(256) void softmax_g(float* __restrict__ sG,
                                                 const int* __restrict__ am) {
  __shared__ float red[256];
  int r = blockIdx.x;  // h*NG + g
  float* row = sG + (long)r * SQ;
  int t = threadIdx.x;
  float v[16], lmax = -1e30f;
#pragma unroll
  for (int u = 0; u < 16; ++u) {
    int s = t + u * 256;
    float x = row[s];
    x = (am[s] != 0) ? x : NEGV;
    v[u] = x;
    lmax = fmaxf(lmax, x);
  }
  float mx = blk_rmax(lmax, red);
  float ls = 0.f;
#pragma unroll
  for (int u = 0; u < 16; ++u) { v[u] = __expf(v[u] - mx); ls += v[u]; }
  float inv = 1.f / blk_rsum(ls, red);
#pragma unroll
  for (int u = 0; u < 16; ++u) row[t + u * 256] = v[u] * inv;
}

// assemble context [S, DM] bf16; rows < NG take the global-query result
__global__ void context_k(const float* __restrict__ attn,
                          const float* __restrict__ outG,
                          __bf16* __restrict__ ctx) {
  long tid = (long)blockIdx.x * blockDim.x + threadIdx.x;
  if (tid >= (long)SQ * DM) return;
  long s = tid / DM;
  int c = (int)(tid % DM), h = c / HD, d = c % HD;
  float v = (s < NG) ? outG[((long)h * NG + s) * HD + d]
                     : attn[((long)h * SQ + s) * HD + d];
  ctx[tid] = (__bf16)v;
}

// classifier
__global__ void cls1_k(const float* __restrict__ x, const float* __restrict__ w1,
                       const float* __restrict__ b1, float* __restrict__ hb) {
  int j = blockIdx.x * blockDim.x + threadIdx.x;
  if (j >= DM) return;
  float acc = b1[j];
  for (int k = 0; k < DM; ++k) acc += x[k] * w1[(long)k * DM + j];
  hb[j] = tanhf(acc);
}
__global__ void cls2_k(const float* __restrict__ hb, const float* __restrict__ w2,
                       const float* __restrict__ b2, float* __restrict__ out) {
  int j = threadIdx.x;
  if (j >= NCLS) return;
  float acc = b2[j];
  for (int k = 0; k < DM; ++k) acc += hb[k] * w2[(long)k * NCLS + j];
  out[j] = acc;
}

// ---------------- host orchestration ----------------
static inline size_t alignup(size_t x) { return (x + 255) & ~(size_t)255; }

extern "C" void kernel_launch(void* const* d_in, const int* in_sizes, int n_in,
                              void* d_out, int out_size, void* d_ws,
                              size_t ws_size, hipStream_t stream) {
  (void)in_sizes; (void)n_in; (void)out_size; (void)ws_size;
  const int*   ids   = (const int*)d_in[0];
  const int*   am    = (const int*)d_in[1];
  const int*   gm    = (const int*)d_in[2];
  const float* emb_w = (const float*)d_in[3];
  const float* emb_p = (const float*)d_in[4];
  const float* ln_es = (const float*)d_in[5];
  const float* ln_eb = (const float*)d_in[6];
  const float* Wq  = (const float*)d_in[7],  *bq  = (const float*)d_in[8];
  const float* Wk  = (const float*)d_in[9],  *bk  = (const float*)d_in[10];
  const float* Wv  = (const float*)d_in[11], *bv  = (const float*)d_in[12];
  const float* Wo  = (const float*)d_in[13], *bo  = (const float*)d_in[14];
  const float* Wqg = (const float*)d_in[15], *bqg = (const float*)d_in[16];
  const float* Wkg = (const float*)d_in[17], *bkg = (const float*)d_in[18];
  const float* Wvg = (const float*)d_in[19], *bvg = (const float*)d_in[20];
  const float* ln1s = (const float*)d_in[21], *ln1b = (const float*)d_in[22];
  const float* ln2s = (const float*)d_in[23], *ln2b = (const float*)d_in[24];
  const float* W1 = (const float*)d_in[25], *b1 = (const float*)d_in[26];
  const float* W2 = (const float*)d_in[27], *b2 = (const float*)d_in[28];
  const float* cw1 = (const float*)d_in[29], *cb1 = (const float*)d_in[30];
  const float* cw2 = (const float*)d_in[31], *cb2 = (const float*)d_in[32];
  float* out = (float*)d_out;

  char* p = (char*)d_ws;
  auto grab = [&](size_t bytes) { void* r = p; p += alignup(bytes); return r; };
  float*  x    = (float*)grab((size_t)SQ * DM * 4);
  __bf16* xb   = (__bf16*)grab((size_t)SQ * DM * 2);
  float*  y    = (float*)grab((size_t)SQ * DM * 4);
  __bf16* wt   = (__bf16*)grab((size_t)DM * DFF * 2);
  __bf16* qb   = (__bf16*)grab((size_t)NH * SQ * HD * 2);
  __bf16* kpad = (__bf16*)grab((size_t)NH * SP * HD * 2);
  __bf16* vt   = (__bf16*)grab((size_t)NH * HD * SP * 2);
  __bf16* qgb  = (__bf16*)grab((size_t)NH * NG * HD * 2);
  __bf16* kgb  = (__bf16*)grab((size_t)NH * SQ * HD * 2);
  __bf16* vgt  = (__bf16*)grab((size_t)NH * HD * SQ * 2);
  float*  band = (float*)grab((size_t)NH * SQ * TW * 4);
  float*  attn = (float*)grab((size_t)NH * SQ * HD * 4);
  float*  sG   = (float*)grab((size_t)NH * NG * SQ * 4);
  float*  outG = (float*)grab((size_t)NH * NG * HD * 4);
  __bf16* ctx  = (__bf16*)grab((size_t)SQ * DM * 2);
  __bf16* h1   = (__bf16*)grab((size_t)SQ * DFF * 2);
  float*  hcls = (float*)grab((size_t)DM * 4);

  const float scl = 0.125f;  // 1/sqrt(HD)

  auto trans = [&](const float* w, int K, int N) {
    long tot = (long)K * N;
    transpose_w<<<(int)((tot + 255) / 256), 256, 0, stream>>>(w, wt, K, N);
  };
  auto gblk = [&](const __bf16* A, const float* bias, int M, int N, int K,
                  int mode, float scale, int gelu, float* Cf,
                  const float* resid, __bf16* Cb, long hs, int rs, int cs,
                  int moff) {
    dim3 g(N / BLK_N, M / BLK_M);
    gemm_act_blk<<<g, 256, 0, stream>>>(A, wt, bias, M, N, K, mode, scale,
                                        gelu, Cf, resid, Cb, hs, rs, cs, moff);
  };
  auto gsm = [&](const __bf16* A, const float* bias, int M, int N, int K,
                 int mode, float scale, int gelu, float* Cf,
                 const float* resid, __bf16* Cb, long hs, int rs, int cs,
                 int moff) {
    dim3 g(N / 16, (M + 127) / 128);
    gemm_act<<<g, 256, 0, stream>>>(A, wt, bias, M, N, K, mode, scale, gelu,
                                    Cf, resid, Cb, hs, rs, cs, moff);
  };

  // embeddings + LN
  {
    long tot = (long)SQ * DM;
    embed_sum<<<(int)((tot + 255) / 256), 256, 0, stream>>>(ids, emb_w, emb_p, y);
    ln_rows<<<SQ, 256, 0, stream>>>(y, ln_es, ln_eb, x, xb);
  }

  for (int l = 0; l < NLAY; ++l) {
    const long wS = (long)l * DM * DM, bS = (long)l * DM;
    zero_bf16_k<<<1024, 256, 0, stream>>>(kpad, (long)NH * SP * HD);
    zero_bf16_k<<<1024, 256, 0, stream>>>(vt, (long)NH * HD * SP);

    // Q,K,V projections -> head-major layouts
    trans(Wq + wS, DM, DM);
    gblk(xb, bq + bS, SQ, DM, DM, 1, scl, 0, nullptr, nullptr, qb,
         (long)SQ * HD, HD, 1, 0);
    trans(Wk + wS, DM, DM);
    gblk(xb, bk + bS, SQ, DM, DM, 1, 1.f, 0, nullptr, nullptr, kpad,
         (long)SP * HD, HD, 1, WW);
    trans(Wv + wS, DM, DM);
    gblk(xb, bv + bS, SQ, DM, DM, 1, 1.f, 0, nullptr, nullptr, vt,
         (long)HD * SP, 1, SP, WW);

    // band scores: per (h,chunk) [W x 3W] = Q[W,HD] * Kpad[3W,HD]^T
    {
      dim3 g(TW / 16, WW / 128, NH * NCH);
      gemm_z<__bf16><<<g, 256, 0, stream>>>(
          qb, kpad, band, WW, TW, HD, HD, HD, TW, NCH,
          (long)SQ * HD, (long)WW * HD, (long)SP * HD, (long)WW * HD,
          (long)SQ * TW, (long)WW * TW, 0);
    }
    band_softmax<<<NH * SQ, 256, 0, stream>>>(qb, kpad, vt, am, gm, band, attn);
    {  // band PV accumulate: attn += P[W,3W] * V3[3W,HD]
      dim3 g(HD / 16, WW / 128, NH * NCH);
      gemm_z<float><<<g, 256, 0, stream>>>(
          band, vt, attn, WW, HD, TW, TW, SP, HD, NCH,
          (long)SQ * TW, (long)WW * TW, (long)HD * SP, (long)WW,
          (long)SQ * HD, (long)WW * HD, 1);
    }

    // global-query full attention with *_global projections
    trans(Wqg + wS, DM, DM);
    gsm(xb, bqg + bS, NG, DM, DM, 1, scl, 0, nullptr, nullptr, qgb,
        (long)NG * HD, HD, 1, 0);
    trans(Wkg + wS, DM, DM);
    gblk(xb, bkg + bS, SQ, DM, DM, 1, 1.f, 0, nullptr, nullptr, kgb,
         (long)SQ * HD, HD, 1, 0);
    trans(Wvg + wS, DM, DM);
    gblk(xb, bvg + bS, SQ, DM, DM, 1, 1.f, 0, nullptr, nullptr, vgt,
         (long)HD * SQ, 1, SQ, 0);
    {  // sG[h,16,S] = Qg * Kg^T
      dim3 g(SQ / 16, 1, NH);
      gemm_z<__bf16><<<g, 256, 0, stream>>>(
          qgb, kgb, sG, NG, SQ, HD, HD, HD, SQ, 1,
          (long)NG * HD, 0, (long)SQ * HD, 0, (long)NG * SQ, 0, 0);
    }
    softmax_g<<<NH * NG, 256, 0, stream>>>(sG, am);
    {  // outG[h,16,HD] = P * Vg
      dim3 g(HD / 16, 1, NH);
      gemm_z<float><<<g, 256, 0, stream>>>(
          sG, vgt, outG, NG, HD, SQ, SQ, SQ, HD, 1,
          (long)NG * SQ, 0, (long)HD * SQ, 0, (long)NG * HD, 0, 0);
    }

    // context assembly + O projection + residual + LN
    {
      long tot = (long)SQ * DM;
      context_k<<<(int)((tot + 255) / 256), 256, 0, stream>>>(attn, outG, ctx);
    }
    trans(Wo + wS, DM, DM);
    gblk(ctx, bo + bS, SQ, DM, DM, 0, 1.f, 0, y, x, nullptr, 0, 0, 0, 0);
    ln_rows<<<SQ, 256, 0, stream>>>(y, ln1s + bS, ln1b + bS, x, xb);

    // FFN
    trans(W1 + (long)l * DM * DFF, DM, DFF);
    gblk(xb, b1 + (long)l * DFF, SQ, DFF, DM, 2, 1.f, 1, nullptr, nullptr, h1,
         0, 0, 0, 0);
    trans(W2 + (long)l * DFF * DM, DFF, DM);
    gblk(h1, b2 + bS, SQ, DM, DFF, 0, 1.f, 0, y, x, nullptr, 0, 0, 0, 0);
    ln_rows<<<SQ, 256, 0, stream>>>(y, ln2s + bS, ln2b + bS, x, xb);
  }

  // classifier head on token 0
  cls1_k<<<3, 256, 0, stream>>>(x, cw1, cb1, hcls);
  cls2_k<<<1, 32, 0, stream>>>(hcls, cw2, cb2, out);
}
